// SwinGraphStage_74534862455397
// MI455X (gfx1250) — compile-verified, matrix operand order checked by reading, not run on hardware
//
#include <hip/hip_runtime.h>
#include <hip/hip_bf16.h>
#include <math.h>

// ---------------------------------------------------------------------------
// MI455X (gfx1250) implementation of the SwinGraphStage reference.
// Dense math: V_WMMA_F32_16X16X32_BF16 (wave32 fragments per CDNA5 ISA 7.12.2).
// GEMM A-tiles staged into LDS with GLOBAL_LOAD_ASYNC_TO_LDS_B128 (ASYNCcnt).
// ---------------------------------------------------------------------------

typedef __bf16 bf16;
typedef __bf16 bf16x16 __attribute__((ext_vector_type(16)));
typedef float  v8f     __attribute__((ext_vector_type(8)));

constexpr int NN    = 32768;   // nodes
constexpr int CC    = 256;     // channels
constexpr int KK    = 128;     // clusters
constexpr int SS    = 256;     // cluster size
constexpr int HH    = 8;       // heads
constexpr int DH    = 64;      // dim head
constexpr int HD    = 512;     // H*DH
constexpr int EE    = 524288;  // edges
constexpr int FF    = 1024;    // ff hidden
constexpr int NKEEP = 16384;   // ceil(0.5*N)

// ---------------------------- WMMA helpers ---------------------------------

__device__ __forceinline__ v8f wmma_bf(bf16x16 a, bf16x16 b, v8f c) {
  return __builtin_amdgcn_wmma_f32_16x16x32_bf16(false, a, false, b,
                                                 (short)0, c, false, false);
}

// A-matrix 16x32 fragment from row-major [*, ld] (ISA 7.12.2, 16-bit A layout).
// Per-lane reads are two contiguous 16B chunks -> ds/global_load_b128.
__device__ __forceinline__ bf16x16 frag_a(const bf16* base, int ld) {
  int l = threadIdx.x & 31;
  int m = l & 15, kh = (l >> 4) << 3;
  const bf16* row = base + m * ld;
  bf16x16 r;
#pragma unroll
  for (int j = 0; j < 8; ++j) r[j] = row[kh + j];
#pragma unroll
  for (int j = 0; j < 8; ++j) r[8 + j] = row[kh + 16 + j];
  return r;
}

// B-matrix 32x16 fragment where B[k][n] = base[n*ld + k] (k-contiguous source:
// transposed LDS tile, K matrix for Q@K^T, or V^T). Also 2x b128 per lane.
__device__ __forceinline__ bf16x16 frag_bT(const bf16* base, int ld) {
  int l = threadIdx.x & 31;
  int n = l & 15, kh = (l >> 4) << 3;
  const bf16* row = base + n * ld;
  bf16x16 r;
#pragma unroll
  for (int j = 0; j < 8; ++j) r[j] = row[kh + j];
#pragma unroll
  for (int j = 0; j < 8; ++j) r[8 + j] = row[kh + 16 + j];
  return r;
}

__device__ __forceinline__ float gelu_f(float x) {
  return 0.5f * x * (1.0f + erff(x * 0.70710678118654752f));
}

// gfx1250 async global->LDS copy (16B per lane), tracked by ASYNCcnt.
__device__ __forceinline__ void async_g2l_b128(unsigned lds_off, const void* g) {
  asm volatile("global_load_async_to_lds_b128 %0, %1, off"
               :: "v"(lds_off), "v"((unsigned long long)(uintptr_t)g)
               : "memory");
}
__device__ __forceinline__ void wait_async0() {
  asm volatile("s_wait_asynccnt 0x0" ::: "memory");
}

// ------------------------------- GEMM --------------------------------------
// out[M,Nd] = act(A[M,Kd] @ W[Kd,Nd] + bias), A/W bf16 row-major.
// Block: 256 threads = 8 waves -> 64(M) x 32(N) tile, K stepped by 32.
// Dynamic LDS: As[64x32] at byte 0 (filled by async b128 copies),
//              Bt[32x32] transposed (n-major) at byte 4096.
// flags: bit0 = GELU, bit1 = bf16 output (else f32).
__global__ __launch_bounds__(256)
void k_gemm(const bf16* __restrict__ A, const bf16* __restrict__ W,
            const float* __restrict__ bias, void* __restrict__ outp,
            int M, int Nd, int Kd, int flags) {
  extern __shared__ char smem[];
  bf16* As = (bf16*)smem;               // LDS byte offset 0
  bf16* Bt = (bf16*)(smem + 4096);      // LDS byte offset 4096
  const int tid = threadIdx.x;
  const int lane = tid & 31, w = tid >> 5;
  const int m0 = blockIdx.y * 64, n0 = blockIdx.x * 32;
  const int mt = (w & 3) * 16, nt = (w >> 2) * 16;
  v8f acc = {};
  for (int kb = 0; kb < Kd; kb += 32) {
    // A tile 64x32: 256 lanes x one async 16B chunk. LDS offset == tid*16.
    {
      int row = tid >> 2, c8 = (tid & 3) * 8;
      async_g2l_b128((unsigned)(tid * 16),
                     A + (size_t)(m0 + row) * Kd + kb + c8);
    }
    // B tile 32x32 -> transposed into Bt[n][k]; vector 16B global reads.
    if (tid < 128) {
      int krow = tid >> 2, c8 = (tid & 3) * 8;
      uint4 pkt = *(const uint4*)(W + (size_t)(kb + krow) * Nd + n0 + c8);
      const bf16* pv = (const bf16*)&pkt;
#pragma unroll
      for (int j = 0; j < 8; ++j) Bt[(c8 + j) * 32 + krow] = pv[j];
    }
    if (kb + 32 < Kd)  // prefetch next W k-slab into L2
      __builtin_prefetch(W + (size_t)(kb + 32 + (tid >> 3)) * Nd + n0, 0, 1);
    wait_async0();
    __syncthreads();
    bf16x16 af = frag_a(&As[mt * 32], 32);
    bf16x16 bfg = frag_bT(&Bt[nt * 32], 32);
    acc = wmma_bf(af, bfg, acc);
    __syncthreads();
  }
  const int nn = lane & 15, mb = (lane >> 4) * 8;
  const int gn = n0 + nt + nn;
  const float bv = bias ? bias[gn] : 0.0f;
#pragma unroll
  for (int j = 0; j < 8; ++j) {
    int gm = m0 + mt + mb + j;
    float v = acc[j] + bv;
    if (flags & 1) v = gelu_f(v);
    if (flags & 2) ((bf16*)outp)[(size_t)gm * Nd + gn] = (bf16)v;
    else           ((float*)outp)[(size_t)gm * Nd + gn] = v;
  }
}

// ------------------------- fused cluster attention --------------------------
// grid (KK, HH, 4); block 128 = 4 waves; wave handles 16 query rows.
// sim = (Q K^T + qv0 + adj*qdv) * scale ; softmax ; out = attn V + v0 + rb*dv
__global__ __launch_bounds__(128)
void k_attn(const bf16* __restrict__ qb, const bf16* __restrict__ kvb,
            const bf16* __restrict__ vt, const unsigned* __restrict__ adj,
            const float* __restrict__ v0dv, bf16* __restrict__ aout) {
  const int cl = blockIdx.x, h = blockIdx.y, qz = blockIdx.z;
  const int tid = threadIdx.x, lane = tid & 31, w = tid >> 5;
  __shared__ bf16 attn_s[4][16 * 256];
  __shared__ unsigned adj_s[64 * 8];
  __shared__ float qv0_s[64], qdv_s[64], rb_s[64];

  const size_t rowbase = (size_t)cl * SS;
  const int i0 = qz * 64;

  for (int e = tid; e < 64 * 8; e += 128)
    adj_s[e] = adj[((size_t)cl * SS + i0 + (e >> 3)) * 8 + (e & 7)];
  if (tid < 64) {
    const bf16* qrow = qb + (rowbase + i0 + tid) * HD + h * DH;
    float s0 = 0.f, s1 = 0.f;
    for (int d = 0; d < DH; ++d) {
      float q = (float)qrow[d];
      s0 += q * v0dv[h * DH + d];
      s1 += q * v0dv[HD + h * DH + d];
    }
    qv0_s[tid] = s0; qdv_s[tid] = s1;
  }
  __syncthreads();

  // pass 1: Q @ K^T, register-resident 16x256 score strip per wave
  const bf16* Abase = qb + (rowbase + i0 + w * 16) * HD + h * DH;
  v8f acc[16];
  const v8f vzero = {};
#pragma unroll
  for (int jt = 0; jt < 16; ++jt) acc[jt] = vzero;
#pragma unroll
  for (int kb = 0; kb < 2; ++kb) {
    bf16x16 af = frag_a(Abase + kb * 32, HD);
#pragma unroll
    for (int jt = 0; jt < 16; ++jt) {
      const bf16* Kb = kvb + (rowbase + jt * 16) * 1024 + h * DH + kb * 32;
      acc[jt] = wmma_bf(af, frag_bT(Kb, 1024), acc[jt]);
    }
  }
  const int nn = lane & 15, mb = (lane >> 4) * 8;
  const float scale = 0.125f;  // DH^-0.5
#pragma unroll
  for (int jt = 0; jt < 16; ++jt) {
#pragma unroll
    for (int j = 0; j < 8; ++j) {
      int lr = w * 16 + mb + j;
      int jc = jt * 16 + nn;
      float ab = ((adj_s[lr * 8 + (jc >> 5)] >> (jc & 31)) & 1u) ? 1.f : 0.f;
      acc[jt][j] = (acc[jt][j] + qv0_s[lr] + ab * qdv_s[lr]) * scale;
    }
  }
  // softmax: row m = mb+j lives on the 16 lanes sharing (lane>>4)
#pragma unroll
  for (int j = 0; j < 8; ++j) {
    int lr = w * 16 + mb + j;
    float mx = -3.0e38f;
#pragma unroll
    for (int jt = 0; jt < 16; ++jt) mx = fmaxf(mx, acc[jt][j]);
    for (int o = 1; o < 16; o <<= 1) mx = fmaxf(mx, __shfl_xor(mx, o, 32));
    float sum = 0.f, sumadj = 0.f;
#pragma unroll
    for (int jt = 0; jt < 16; ++jt) {
      int jc = jt * 16 + nn;
      float e = __expf(acc[jt][j] - mx);
      acc[jt][j] = e;
      sum += e;
      if ((adj_s[lr * 8 + (jc >> 5)] >> (jc & 31)) & 1u) sumadj += e;
    }
    for (int o = 1; o < 16; o <<= 1) {
      sum += __shfl_xor(sum, o, 32);
      sumadj += __shfl_xor(sumadj, o, 32);
    }
    float inv = 1.0f / sum;
#pragma unroll
    for (int jt = 0; jt < 16; ++jt)
      attn_s[w][(mb + j) * 256 + jt * 16 + nn] = (bf16)(acc[jt][j] * inv);
    if (nn == 0) rb_s[lr] = sumadj * inv;
  }
  __syncthreads();

  // pass 2: attn @ V  (attn bf16 in LDS as A; V^T streamed: k-contiguous b128)
#pragma unroll
  for (int ntile = 0; ntile < 4; ++ntile) {
    v8f o2 = vzero;
#pragma unroll
    for (int kb2 = 0; kb2 < 8; ++kb2) {
      bf16x16 af2 = frag_a(&attn_s[w][kb2 * 32], 256);
      const bf16* Vb = vt + (size_t)(h * DH + ntile * 16) * NN
                          + rowbase + kb2 * 32;
      o2 = wmma_bf(af2, frag_bT(Vb, NN), o2);
    }
    int dcol = h * DH + ntile * 16 + nn;
#pragma unroll
    for (int j = 0; j < 8; ++j) {
      int lr = w * 16 + mb + j;
      float v = o2[j] + v0dv[dcol] + rb_s[lr] * v0dv[HD + dcol];
      aout[(rowbase + i0 + lr) * HD + dcol] = (bf16)v;
    }
  }
}

// V^T materialization: vt[d][r] = kv[r][512 + d]
__global__ __launch_bounds__(256)
void k_vt(const bf16* __restrict__ kv, bf16* __restrict__ vt) {
  int r = blockIdx.x, c = threadIdx.x;
#pragma unroll
  for (int t = 0; t < 2; ++t) {
    int d = c + t * 256;
    vt[(size_t)d * NN + r] = kv[(size_t)r * 1024 + HD + d];
  }
}

// --------------------------- elementwise kernels ----------------------------

__global__ __launch_bounds__(256)
void k_ln(const float* __restrict__ x, const int* __restrict__ idx,
          const float* __restrict__ g, const float* __restrict__ b,
          bf16* __restrict__ out) {
  int r = blockIdx.x, c = threadIdx.x;
  int src = idx ? idx[r] : r;
  float v = x[(size_t)src * CC + c];
  __shared__ float red[256];
  red[c] = v; __syncthreads();
  for (int o = 128; o > 0; o >>= 1) { if (c < o) red[c] += red[c + o]; __syncthreads(); }
  float mean = red[0] * (1.0f / CC); __syncthreads();
  float d = v - mean;
  red[c] = d * d; __syncthreads();
  for (int o = 128; o > 0; o >>= 1) { if (c < o) red[c] += red[c + o]; __syncthreads(); }
  float var = red[0] * (1.0f / CC);
  out[(size_t)r * CC + c] = (bf16)(d * rsqrtf(var + 1e-5f) * g[c] + b[c]);
}

__global__ __launch_bounds__(256)
void k_rope(bf16* __restrict__ q, bf16* __restrict__ kv) {
  int r = blockIdx.x;          // row in cluster order
  int p = threadIdx.x;         // pair index over 512 dims
  int pos = r & (SS - 1);
  int t = p & 31;              // pair within head
  float inv = __powf(10000.f, -(float)(2 * t) / 64.f);
  float sn, cs;
  __sincosf(pos * inv, &sn, &cs);
  size_t qi = (size_t)r * HD + 2 * p;
  float q0 = (float)q[qi], q1 = (float)q[qi + 1];
  q[qi]     = (bf16)(q0 * cs - q1 * sn);
  q[qi + 1] = (bf16)(q1 * cs + q0 * sn);
  size_t ki = (size_t)r * 1024 + 2 * p;  // K = cols 0..511 of kv
  float k0 = (float)kv[ki], k1 = (float)kv[ki + 1];
  kv[ki]     = (bf16)(k0 * cs - k1 * sn);
  kv[ki + 1] = (bf16)(k1 * cs + k0 * sn);
}

// gated residual: gate = sigmoid([x, res, x-res] @ gw); out = x*g + res*(1-g)
__global__ __launch_bounds__(256)
void k_gate(const float* __restrict__ xin, const float* __restrict__ resb,
            const int* __restrict__ ridx, const float* __restrict__ gw,
            float* __restrict__ out) {
  int r = blockIdx.x, c = threadIdx.x;
  size_t i = (size_t)r * CC + c;
  float xv = xin[i];
  size_t rrow = ridx ? (size_t)ridx[r] : (size_t)r;
  float rv = resb[rrow * CC + c];
  __shared__ float red[256];
  red[c] = xv * gw[c] + rv * gw[256 + c] + (xv - rv) * gw[512 + c];
  __syncthreads();
  for (int o = 128; o > 0; o >>= 1) { if (c < o) red[c] += red[c + o]; __syncthreads(); }
  float gt = 1.0f / (1.0f + __expf(-red[0]));
  out[i] = xv * gt + rv * (1.0f - gt);
}

__global__ void k_scatter_add(float* __restrict__ x, const int* __restrict__ idx,
                              const float* __restrict__ nodes) {
  int r = blockIdx.x, c = threadIdx.x;
  x[(size_t)idx[r] * CC + c] += nodes[(size_t)r * CC + c];
}

__global__ void k_add(float* __restrict__ x, const float* __restrict__ u) {
  size_t i = (size_t)blockIdx.x * 256 + threadIdx.x;
  x[i] += u[i];
}

__global__ void k_cvt(const float* __restrict__ a, bf16* __restrict__ o, int n) {
  int i = blockIdx.x * 256 + threadIdx.x;
  if (i < n) o[i] = (bf16)a[i];
}

// e01 = adj_emb @ We + be ; store v0 = e01[0], dv = e01[1]-e01[0]
__global__ __launch_bounds__(256)
void k_e01(const float* __restrict__ emb, const float* __restrict__ We,
           const float* __restrict__ be, float* __restrict__ v0dv) {
  int c = blockIdx.x * 256 + threadIdx.x;
  if (c >= HD) return;
  float s0 = be[c], s1 = be[c];
  for (int k2 = 0; k2 < CC; ++k2) {
    float wv = We[k2 * HD + c];
    s0 += emb[k2] * wv;
    s1 += emb[CC + k2] * wv;
  }
  v0dv[c] = s0;
  v0dv[HD + c] = s1 - s0;
}

// -------------------------- adjacency bitmask -------------------------------

__global__ void k_nodemaps(const int* __restrict__ cl, int* __restrict__ nc,
                           int* __restrict__ np) {
  int r = blockIdx.x * 256 + threadIdx.x;
  if (r < NN) { int g = cl[r]; nc[g] = r >> 8; np[g] = r & 255; }
}

__global__ void k_adjedges(const int* __restrict__ ei, const int* __restrict__ nc,
                           const int* __restrict__ np, unsigned* __restrict__ adj) {
  int e = blockIdx.x * 256 + threadIdx.x;
  if (e >= EE) return;
  int u = ei[e], v2 = ei[EE + e];
  int cu = nc[u], cv = nc[v2];
  if (cu != cv) return;
  int pu = np[u], pv = np[v2];
  atomicOr(&adj[((size_t)cu * SS + pu) * 8 + (pv >> 5)], 1u << (pv & 31));
}

// ------------------------------- pooling ------------------------------------

__global__ void k_wnorm(const float* __restrict__ w, float* __restrict__ invn) {
  __shared__ float red[256];
  int c = threadIdx.x;
  float v = w[c];
  red[c] = v * v; __syncthreads();
  for (int o = 128; o > 0; o >>= 1) { if (c < o) red[c] += red[c + o]; __syncthreads(); }
  if (c == 0) invn[0] = rsqrtf(red[0]);
}

__global__ void k_score(const float* __restrict__ xe, const float* __restrict__ w,
                        const float* __restrict__ invn, float* __restrict__ sc,
                        int* __restrict__ si) {
  int r = blockIdx.x, c = threadIdx.x;
  __shared__ float red[256];
  red[c] = xe[(size_t)r * CC + c] * w[c]; __syncthreads();
  for (int o = 128; o > 0; o >>= 1) { if (c < o) red[c] += red[c + o]; __syncthreads(); }
  if (c == 0) { sc[r] = tanhf(red[0] * invn[0]); si[r] = r; }
}

__global__ void k_bitonic(float* __restrict__ sc, int* __restrict__ si,
                          int j, int kstep) {
  int i = blockIdx.x * 256 + threadIdx.x;
  int ixj = i ^ j;
  if (ixj <= i) return;
  bool up = (i & kstep) == 0;                  // "up" => better element first
  float a = sc[i], b2 = sc[ixj];
  int ia = si[i], ib = si[ixj];
  bool aBetter = (a > b2) || (a == b2 && ia < ib);
  if (up ? !aBetter : aBetter) {
    sc[i] = b2; sc[ixj] = a; si[i] = ib; si[ixj] = ia;
  }
}

__global__ void k_pool_out(const float* __restrict__ xe, const float* __restrict__ sc,
                           const int* __restrict__ si, float* __restrict__ out) {
  int m = blockIdx.x, c = threadIdx.x;
  float v = sc[m]; int p = si[m];
  out[(size_t)m * CC + c] = xe[(size_t)p * CC + c] * v;
  if (c == 0) {
    ((int*)out)[(size_t)NKEEP * CC + m] = p;            // perm (int bits)
    out[(size_t)NKEEP * CC + NKEEP + m] = v;            // vals
  }
}

// ------------------------------- host side ----------------------------------
// d_in layout (top-level in setup_inputs() insertion order; nested `params`
// pytree flattened in sorted-key order):
//  0 x  1 edge_index  2 clusters1  3 clusters2
//  4 expand.b  5 expand.w
//  6 mlp1.l1.b 7 mlp1.l1.w 8 mlp1.l2.b 9 mlp1.l2.w
// 10 mlp2.l1.b 11 mlp2.l1.w 12 mlp2.l2.b 13 mlp2.l2.w
// 14 pool_w
// 15..33 shift block, 34..52 window block, each block sorted:
//  +0 adj_emb +1 edges_to_kv.b +2 edges_to_kv.w +3 ff1.b +4 ff1.w +5 ff2.b
//  +6 ff2.w +7 gate_attn +8 gate_ff +9 ln1_b +10 ln1_g +11 ln2_b +12 ln2_g
//  +13 to_kv.b +14 to_kv.w +15 to_out.b +16 to_out.w +17 to_q.b +18 to_q.w

struct BlockP {
  const float *adj_emb, *ekv_b, *ekv_w, *ff1_b, *ff1_w, *ff2_b, *ff2_w,
              *gate_attn, *gate_ff, *ln1_b, *ln1_g, *ln2_b, *ln2_g,
              *kv_b, *kv_w, *o_b, *o_w, *q_b, *q_w;
};
struct BlockWB { bf16 *q_w, *kv_w, *o_w, *ff1_w, *ff2_w; };

static BlockP make_block(void* const* d_in, int base) {
  BlockP b;
  const float** f = (const float**)&b;
  for (int i = 0; i < 19; ++i) f[i] = (const float*)d_in[base + i];
  return b;
}

extern "C" void kernel_launch(void* const* d_in, const int* in_sizes, int n_in,
                              void* d_out, int out_size, void* d_ws, size_t ws_size,
                              hipStream_t stream) {
  (void)in_sizes; (void)n_in; (void)out_size; (void)ws_size;

  const float* x_in      = (const float*)d_in[0];
  const int*   edge_ix   = (const int*)d_in[1];
  const int*   clusters1 = (const int*)d_in[2];
  const int*   clusters2 = (const int*)d_in[3];
  const float* expand_b  = (const float*)d_in[4];
  const float* expand_w  = (const float*)d_in[5];
  const float* m1l1_b = (const float*)d_in[6],  *m1l1_w = (const float*)d_in[7];
  const float* m1l2_b = (const float*)d_in[8],  *m1l2_w = (const float*)d_in[9];
  const float* m2l1_b = (const float*)d_in[10], *m2l1_w = (const float*)d_in[11];
  const float* m2l2_b = (const float*)d_in[12], *m2l2_w = (const float*)d_in[13];
  const float* pool_w = (const float*)d_in[14];
  BlockP shiftB  = make_block(d_in, 15);
  BlockP windowB = make_block(d_in, 34);

  // ---- workspace carve-out ----
  char* cur = (char*)d_ws;
  auto carve = [&](size_t bytes) {
    char* r = cur; cur += (bytes + 255) & ~(size_t)255; return r;
  };
  float*    xbuf  = (float*)   carve((size_t)NN * CC * 4);
  bf16*     hbuf  = (bf16*)    carve((size_t)NN * CC * 2);
  bf16*     qb    = (bf16*)    carve((size_t)NN * HD * 2);
  bf16*     kvb   = (bf16*)    carve((size_t)NN * 1024 * 2);
  bf16*     vtb   = (bf16*)    carve((size_t)NN * HD * 2);
  bf16*     aob   = (bf16*)    carve((size_t)NN * HD * 2);
  float*    obuf  = (float*)   carve((size_t)NN * CC * 4);
  float*    nodes = (float*)   carve((size_t)NN * CC * 4);
  bf16*     ffb   = (bf16*)    carve((size_t)NN * FF * 2);
  unsigned* adjm  = (unsigned*)carve((size_t)KK * SS * 8 * 4);
  int*      ncm   = (int*)     carve((size_t)NN * 4);
  int*      npm   = (int*)     carve((size_t)NN * 4);
  float*    v0dv  = (float*)   carve(1024 * 4);
  bf16*     wbp   = (bf16*)    carve((size_t)2424832 * 2);
  float*    scb   = (float*)   carve((size_t)NN * 4);
  int*      sib   = (int*)     carve((size_t)NN * 4);
  float*    invn  = (float*)   carve(256);

  // bf16 weight pool layout
  BlockWB wbW, wbS;
  size_t wo = 0;
  auto takew = [&](size_t n) { bf16* r = wbp + wo; wo += n; return r; };
  wbW.q_w = takew(256 * 512); wbW.kv_w = takew(256 * 1024); wbW.o_w = takew(512 * 256);
  wbW.ff1_w = takew(256 * 1024); wbW.ff2_w = takew(1024 * 256);
  wbS.q_w = takew(256 * 512); wbS.kv_w = takew(256 * 1024); wbS.o_w = takew(512 * 256);
  wbS.ff1_w = takew(256 * 1024); wbS.ff2_w = takew(1024 * 256);
  bf16* wm1l1 = takew(256 * 256); bf16* wm1l2 = takew(256 * 256);
  bf16* wm2l1 = takew(256 * 256); bf16* wm2l2 = takew(256 * 256);
  bf16* wexp  = takew(256 * 256);

  auto cvt = [&](const float* s, bf16* d, int n) {
    k_cvt<<<(n + 255) / 256, 256, 0, stream>>>(s, d, n);
  };
  auto gemm = [&](const bf16* A, const bf16* W, const float* bias, void* out,
                  int M, int Nd, int Kd, int flags) {
    dim3 grid(Nd / 32, M / 64);
    k_gemm<<<grid, 256, 6144, stream>>>(A, W, bias, out, M, Nd, Kd, flags);
  };

  // ---- convert weights to bf16 (every launch; deterministic) ----
  cvt(windowB.q_w, wbW.q_w, 256 * 512);   cvt(windowB.kv_w, wbW.kv_w, 256 * 1024);
  cvt(windowB.o_w, wbW.o_w, 512 * 256);   cvt(windowB.ff1_w, wbW.ff1_w, 256 * 1024);
  cvt(windowB.ff2_w, wbW.ff2_w, 1024 * 256);
  cvt(shiftB.q_w, wbS.q_w, 256 * 512);    cvt(shiftB.kv_w, wbS.kv_w, 256 * 1024);
  cvt(shiftB.o_w, wbS.o_w, 512 * 256);    cvt(shiftB.ff1_w, wbS.ff1_w, 256 * 1024);
  cvt(shiftB.ff2_w, wbS.ff2_w, 1024 * 256);
  cvt(m1l1_w, wm1l1, 256 * 256); cvt(m1l2_w, wm1l2, 256 * 256);
  cvt(m2l1_w, wm2l1, 256 * 256); cvt(m2l2_w, wm2l2, 256 * 256);
  cvt(expand_w, wexp, 256 * 256);

  // running features
  hipMemcpyAsync(xbuf, x_in, (size_t)NN * CC * 4, hipMemcpyDeviceToDevice, stream);

  auto run_block = [&](const BlockP& P, const BlockWB& W, const int* clusters) {
    // adjacency bitmask for this clustering
    hipMemsetAsync(adjm, 0, (size_t)KK * SS * 8 * 4, stream);
    k_nodemaps<<<NN / 256, 256, 0, stream>>>(clusters, ncm, npm);
    k_adjedges<<<EE / 256, 256, 0, stream>>>(edge_ix, ncm, npm, adjm);
    // edge kv folded terms
    k_e01<<<2, 256, 0, stream>>>(P.adj_emb, P.ekv_w, P.ekv_b, v0dv);
    // prenorm (gathered into cluster order)
    k_ln<<<NN, 256, 0, stream>>>(xbuf, clusters, P.ln1_g, P.ln1_b, hbuf);
    // projections
    gemm(hbuf, W.q_w, P.q_b, qb, NN, HD, CC, /*bf16*/2);
    gemm(hbuf, W.kv_w, P.kv_b, kvb, NN, 1024, CC, 2);
    // rotary on q and k, then materialize V^T
    k_rope<<<NN, 256, 0, stream>>>(qb, kvb);
    k_vt<<<NN, 256, 0, stream>>>(kvb, vtb);
    // fused adjacency-biased attention
    k_attn<<<dim3(KK, HH, 4), 128, 0, stream>>>(qb, kvb, vtb, adjm, v0dv, aob);
    // out projection + gated residual vs gathered x
    gemm(aob, W.o_w, P.o_b, obuf, NN, CC, HD, 0);
    k_gate<<<NN, 256, 0, stream>>>(obuf, xbuf, clusters, P.gate_attn, nodes);
    // FF with prenorm + gated residual
    k_ln<<<NN, 256, 0, stream>>>(nodes, nullptr, P.ln2_g, P.ln2_b, hbuf);
    gemm(hbuf, W.ff1_w, P.ff1_b, ffb, NN, FF, CC, /*gelu+bf16*/3);
    gemm(ffb, W.ff2_w, P.ff2_b, obuf, NN, CC, FF, 0);
    k_gate<<<NN, 256, 0, stream>>>(obuf, nodes, nullptr, P.gate_ff, nodes);
    // x[clusters] += nodes
    k_scatter_add<<<NN, 256, 0, stream>>>(xbuf, clusters, nodes);
  };

  auto run_mlp = [&](const bf16* w1, const float* b1, const bf16* w2,
                     const float* b2) {
    cvt(xbuf, hbuf, NN * CC);
    gemm(hbuf, w1, b1, ffb, NN, CC, CC, 3);      // gelu, bf16 (uses first N*C of ffb)
    gemm(ffb, w2, b2, obuf, NN, CC, CC, 0);
    k_add<<<(NN * CC) / 256, 256, 0, stream>>>(xbuf, obuf);
  };

  // ---- stage ----
  run_block(windowB, wbW, clusters1);
  run_mlp(wm1l1, m1l1_b, wm1l2, m1l2_b);
  run_block(shiftB, wbS, clusters2);
  run_mlp(wm2l1, m2l1_b, wm2l2, m2l2_b);

  // expand
  cvt(xbuf, hbuf, NN * CC);
  gemm(hbuf, wexp, expand_b, obuf, NN, CC, CC, 0);   // obuf = expanded x

  // top-k pooling
  k_wnorm<<<1, 256, 0, stream>>>(pool_w, invn);
  k_score<<<NN, 256, 0, stream>>>(obuf, pool_w, invn, scb, sib);
  for (int kstep = 2; kstep <= NN; kstep <<= 1)
    for (int j = kstep >> 1; j > 0; j >>= 1)
      k_bitonic<<<NN / 256, 256, 0, stream>>>(scb, sib, j, kstep);
  k_pool_out<<<NKEEP, 256, 0, stream>>>(obuf, scb, sib, (float*)d_out);
}
